// AutoregressivePolicy_62577673503211
// MI455X (gfx1250) — matrix operand order repaired and checked by reading, not compiled, not run on hardware
//
#include <hip/hip_runtime.h>

// ---------------------------------------------------------------------------
// Types / helpers
// ---------------------------------------------------------------------------
typedef __bf16 bf16_t;
typedef bf16_t v16bf __attribute__((ext_vector_type(16)));
typedef bf16_t v8bf  __attribute__((ext_vector_type(8)));
typedef float  v8f   __attribute__((ext_vector_type(8)));

__device__ __forceinline__ bf16_t f2bf(float f) {
    unsigned u = __builtin_bit_cast(unsigned, f);
    unsigned short h = (unsigned short)(u >> 16);
    return __builtin_bit_cast(bf16_t, h);
}
__device__ __forceinline__ float bf2f(bf16_t b) {
    unsigned short h = __builtin_bit_cast(unsigned short, b);
    unsigned u = ((unsigned)h) << 16;
    return __builtin_bit_cast(float, u);
}

__device__ __forceinline__ v8f wmma_bf16(v16bf a, v16bf b, v8f c) {
    return __builtin_amdgcn_wmma_f32_16x16x32_bf16(
        false, a, false, b, (short)0, c, false, false);
}

__device__ __forceinline__ v8f v8f_zero() {
    v8f c;
#pragma unroll
    for (int i = 0; i < 8; ++i) c[i] = 0.0f;
    return c;
}

// A-operand (16x32 bf16) from LDS. contiguous k layout (same as B loaders):
//   lanes 0-15 : M=lane,    halves 0..15 -> K=0..15
//   lanes 16-31: M=lane-16, halves 0..15 -> K=16..31
__device__ __forceinline__ v16bf load_a_lds(const bf16_t* base, int ld, int lane, int k0) {
    int m = lane & 15, g = (lane >> 4) & 1;
    const bf16_t* p = base + m * ld + k0 + g * 16;
    v8bf lo = *(const v8bf*)(p);
    v8bf hi = *(const v8bf*)(p + 8);
    v16bf a;
#pragma unroll
    for (int h = 0; h < 8; ++h) { a[h] = lo[h]; a[h + 8] = hi[h]; }
    return a;
}

// A-operand from global fp32 X via b128 loads (no guards; ldx mult of 32).
__device__ __forceinline__ v16bf load_a_g4(const float* X, int ldx, int m0, int k0, int lane) {
    int m = lane & 15, g = (lane >> 4) & 1;
    const float* p = X + (size_t)(m0 + m) * ldx + k0 + g * 16;
    float4 a0 = ((const float4*)p)[0];
    float4 a1 = ((const float4*)p)[1];
    float4 b0 = ((const float4*)(p + 8))[0];
    float4 b1 = ((const float4*)(p + 8))[1];
    v16bf a;
    a[0] = f2bf(a0.x); a[1] = f2bf(a0.y); a[2] = f2bf(a0.z); a[3] = f2bf(a0.w);
    a[4] = f2bf(a1.x); a[5] = f2bf(a1.y); a[6] = f2bf(a1.z); a[7] = f2bf(a1.w);
    a[8] = f2bf(b0.x); a[9] = f2bf(b0.y); a[10] = f2bf(b0.z); a[11] = f2bf(b0.w);
    a[12] = f2bf(b1.x); a[13] = f2bf(b1.y); a[14] = f2bf(b1.z); a[15] = f2bf(b1.w);
    return a;
}

// B-operand (32x16 bf16, B[k][n] = W[n][k]) from global fp32 W (N, ldw),
// b128 loads, no guards (ldw compile-time multiple of 32 at call sites).
// Contiguous k layout (same as A loaders): lane -> n=lane&15, k = (lane>>4)*16 + h.
__device__ __forceinline__ v16bf load_b_w4(const float* W, int ldw, int n0, int k0, int lane) {
    int n = lane & 15, g = (lane >> 4) & 1;
    const float* p = W + (size_t)(n0 + n) * ldw + k0 + g * 16;
    float4 q0 = ((const float4*)p)[0];
    float4 q1 = ((const float4*)p)[1];
    float4 q2 = ((const float4*)p)[2];
    float4 q3 = ((const float4*)p)[3];
    v16bf b;
    b[0] = f2bf(q0.x); b[1] = f2bf(q0.y); b[2] = f2bf(q0.z); b[3] = f2bf(q0.w);
    b[4] = f2bf(q1.x); b[5] = f2bf(q1.y); b[6] = f2bf(q1.z); b[7] = f2bf(q1.w);
    b[8] = f2bf(q2.x); b[9] = f2bf(q2.y); b[10] = f2bf(q2.z); b[11] = f2bf(q2.w);
    b[12] = f2bf(q3.x); b[13] = f2bf(q3.y); b[14] = f2bf(q3.z); b[15] = f2bf(q3.w);
    return b;
}

// B-operand from bf16 LDS (zero-padded), contiguous 16 bf16 per lane.
__device__ __forceinline__ v16bf load_b_lds(const bf16_t* W, int ld, int n0, int k0, int lane) {
    int n = lane & 15, g = (lane >> 4) & 1;
    const bf16_t* p = W + (n0 + n) * ld + k0 + g * 16;
    v8bf lo = *(const v8bf*)(p);
    v8bf hi = *(const v8bf*)(p + 8);
    v16bf b;
#pragma unroll
    for (int h = 0; h < 8; ++h) { b[h] = lo[h]; b[h + 8] = hi[h]; }
    return b;
}

// ---------------------------------------------------------------------------
// Constants
// ---------------------------------------------------------------------------
#define BATCH   64
#define TSTEPS  80
#define DMODEL  256
#define NTOK    392      // 64 agents + 256 lanes + 8 route + 64 poly
#define SEG_AG  0
#define SEG_LN  64
#define SEG_RT  320
#define SEG_PL  328

// ---------------------------------------------------------------------------
// Fused PointNet encoder (WMMA), maxpool over group, * mask -> feats_all.
// G==1 => no pooling (agents), block handles 32 points; else block per group.
// ---------------------------------------------------------------------------
__global__ void enc_kernel(const float* __restrict__ X, const float* __restrict__ mask,
                           const float* __restrict__ W1, const float* __restrict__ b1,
                           const float* __restrict__ W2, const float* __restrict__ b2,
                           const float* __restrict__ W3, const float* __restrict__ b3,
                           float* __restrict__ feats_all,
                           int din, int G, int npb, int seg_off) {
    __shared__ __align__(16) bf16_t Xs[32 * 32];
    __shared__ __align__(16) bf16_t W1s[64 * 32];
    __shared__ __align__(16) bf16_t H1[32 * 64];
    __shared__ __align__(16) bf16_t H2[32 * 128];
    __shared__ __align__(16) float  OutF[32 * 256];

    const int tid = threadIdx.x, lane = tid & 31, wave = tid >> 5;
    int rows;
    size_t xrow0;
    if (G == 1) { rows = 32; xrow0 = (size_t)blockIdx.x * 32; }
    else        { rows = G;  xrow0 = (size_t)blockIdx.x * G;  }

    // stage input tile (zero-padded to 32x32) and W1 (zero-padded to 64x32)
    for (int i = tid; i < 32 * 32; i += 256) {
        int r = i >> 5, k = i & 31;
        float v = (r < rows && k < din) ? X[(xrow0 + r) * din + k] : 0.0f;
        Xs[i] = f2bf(v);
    }
    for (int i = tid; i < 64 * 32; i += 256) {
        int r = i >> 5, k = i & 31;
        W1s[i] = f2bf(k < din ? W1[r * din + k] : 0.0f);
    }
    __syncthreads();

    // ---- Layer 1: (32x32) @ W1(64x32)^T -> H1 (32x64), relu ----
    {
        int mt = wave >> 2, nt = wave & 3;                 // 2 x 4 tiles
        v8f c = v8f_zero();
        v16bf a = load_a_lds(Xs + mt * 16 * 32, 32, lane, 0);
        v16bf b = load_b_lds(W1s, 32, nt * 16, 0, lane);
        c = wmma_bf16(a, b, c);
        int n = lane & 15, mb = (lane >> 4) * 8;
        float bs = b1[nt * 16 + n];
#pragma unroll
        for (int r = 0; r < 8; ++r) {
            float v = c[r] + bs;
            H1[(mt * 16 + mb + r) * 64 + nt * 16 + n] = f2bf(v > 0.0f ? v : 0.0f);
        }
    }
    __syncthreads();

    // ---- Layer 2: (32x64) @ W2(128,64)^T -> H2 (32x128), relu ----
    for (int t = wave; t < 16; t += 8) {                   // 2 x 8 tiles
        int mt = t >> 3, nt = t & 7;
        v8f c = v8f_zero();
#pragma unroll
        for (int k0 = 0; k0 < 64; k0 += 32) {
            v16bf a = load_a_lds(H1 + mt * 16 * 64, 64, lane, k0);
            v16bf b = load_b_w4(W2, 64, nt * 16, k0, lane);
            c = wmma_bf16(a, b, c);
        }
        int n = lane & 15, mb = (lane >> 4) * 8;
        float bs = b2[nt * 16 + n];
#pragma unroll
        for (int r = 0; r < 8; ++r) {
            float v = c[r] + bs;
            H2[(mt * 16 + mb + r) * 128 + nt * 16 + n] = f2bf(v > 0.0f ? v : 0.0f);
        }
    }
    __syncthreads();

    // ---- Layer 3: (32x128) @ W3(256,128)^T -> OutF (32x256), +bias ----
    for (int t = wave; t < 32; t += 8) {                   // 2 x 16 tiles
        int mt = t >> 4, nt = t & 15;
        v8f c = v8f_zero();
#pragma unroll
        for (int k0 = 0; k0 < 128; k0 += 32) {
            v16bf a = load_a_lds(H2 + mt * 16 * 128, 128, lane, k0);
            v16bf b = load_b_w4(W3, 128, nt * 16, k0, lane);
            c = wmma_bf16(a, b, c);
        }
        int n = lane & 15, mb = (lane >> 4) * 8;
        float bs = b3[nt * 16 + n];
#pragma unroll
        for (int r = 0; r < 8; ++r)
            OutF[(mt * 16 + mb + r) * 256 + nt * 16 + n] = c[r] + bs;
    }
    __syncthreads();

    // ---- pool / scatter ----
    if (G == 1) {
        for (int i = tid; i < 32 * 256; i += 256) {
            int r = i >> 8, d = i & 255;
            int grp = (int)blockIdx.x * 32 + r;
            int b = grp / npb, j = grp % npb;
            feats_all[((size_t)b * NTOK + seg_off + j) * DMODEL + d] = OutF[i] * mask[grp];
        }
    } else {
        int grp = (int)blockIdx.x;
        int b = grp / npb, j = grp % npb;
        float mk = mask[grp];
        int d = tid;
        float mx = OutF[d];
        for (int r = 1; r < G; ++r) mx = fmaxf(mx, OutF[r * 256 + d]);
        feats_all[((size_t)b * NTOK + seg_off + j) * DMODEL + d] = mx * mk;
    }
}

// ---------------------------------------------------------------------------
// WMMA GEMM (K=N=256 fixed): out(M,256) = X(M,256) @ W(256,256)^T + bias, bf16 out
// ---------------------------------------------------------------------------
__global__ void gemm_xwt_kernel(const float* __restrict__ X, const float* __restrict__ W,
                                const float* __restrict__ bias, bf16_t* __restrict__ out,
                                int M) {
    const int tid = threadIdx.x, lane = tid & 31, wave = tid >> 5;
    const int nt_total = DMODEL / 16;
    int tile = (int)blockIdx.x * 8 + wave;
    if (tile >= (M >> 4) * nt_total) return;
    int mt = tile / nt_total, nt = tile % nt_total;
    v8f c = v8f_zero();
#pragma unroll
    for (int k0 = 0; k0 < DMODEL; k0 += 32) {
        v16bf a = load_a_g4(X, DMODEL, mt * 16, k0, lane);
        v16bf b = load_b_w4(W, DMODEL, nt * 16, k0, lane);
        c = wmma_bf16(a, b, c);
    }
    int n = lane & 15, mb = (lane >> 4) * 8;
    float bs = bias[nt * 16 + n];
#pragma unroll
    for (int r = 0; r < 8; ++r)
        out[(size_t)(mt * 16 + mb + r) * DMODEL + nt * 16 + n] = f2bf(c[r] + bs);
}

// ---------------------------------------------------------------------------
// fp32 -> bf16 weight conversion
// ---------------------------------------------------------------------------
__global__ void cvt_bf16_kernel(const float* __restrict__ src, bf16_t* __restrict__ dst, int n) {
    int i = ((int)blockIdx.x * 256 + (int)threadIdx.x) * 4;
    if (i + 3 < n) {
        float4 v = *(const float4*)(src + i);
        dst[i + 0] = f2bf(v.x); dst[i + 1] = f2bf(v.y);
        dst[i + 2] = f2bf(v.z); dst[i + 3] = f2bf(v.w);
    }
}

// ---------------------------------------------------------------------------
// query0: lon half of the concat is a constant => scaled row-sum of combine_w.
// ---------------------------------------------------------------------------
__global__ void query0_kernel(const float* __restrict__ feats_all,
                              const float* __restrict__ combine_w,
                              const float* __restrict__ combine_b,
                              const int* __restrict__ mode_c,
                              float* __restrict__ q0) {
    int b = blockIdx.x, o = threadIdx.x;
    int mc = mode_c[b];
    int lon = mc / 8, lat = mc % 8;
    float lv = (float)lon / 12.0f;
    const float* rf = feats_all + ((size_t)b * NTOK + SEG_RT + lat) * DMODEL;
    const float* w = combine_w + (size_t)o * 512;
    float s = combine_b[o], ws = 0.0f;
    for (int d = 0; d < 256; ++d) { ws += w[d]; s += rf[d] * w[256 + d]; }
    q0[b * DMODEL + o] = s + lv * ws;
}

// ---------------------------------------------------------------------------
// Decode: one block per batch element; wave-per-row bf16 GEMVs.
// ---------------------------------------------------------------------------
struct DecArgs {
    const float* agent_poses; const float* map_poses; const float* poly_poses;
    const float* q0;
    const bf16_t* Kl[3]; const bf16_t* Vl[3];
    const bf16_t* wq[3]; const bf16_t* wo[3];      // (256,256) bf16
    const bf16_t* f1w[3]; const bf16_t* f2w[3];    // (1024,256) / (256,1024) bf16
    const float* in_b[3]; const float* out_b[3];
    const float* g1[3]; const float* bb1[3];
    const float* fb1[3]; const float* fb2[3];
    const float* g2[3]; const float* bb2[3];
    const float* act_w; const float* act_b;
    float* out;
};

__device__ __forceinline__ float wave_reduce(float s) {
#pragma unroll
    for (int m = 16; m > 0; m >>= 1) s += __shfl_xor(s, m, 32);
    return s;
}

// out[row] = (relu?) dot(W[row,0:256], xin) + bias[row], rows = wave*rpw..+rpw
__device__ __forceinline__ void wave_gemv256(const bf16_t* __restrict__ W,
                                             const float* __restrict__ bias,
                                             const float* __restrict__ xin,
                                             float* __restrict__ out,
                                             int rpw, int wave, int lane, bool relu) {
    float x8[8];
#pragma unroll
    for (int e = 0; e < 8; ++e) x8[e] = xin[lane * 8 + e];
    int r0 = wave * rpw;
    for (int r = 0; r < rpw; ++r) {
        int row = r0 + r;
        v8bf w = ((const v8bf*)(W + (size_t)row * 256))[lane];
        float s = 0.0f;
#pragma unroll
        for (int e = 0; e < 8; ++e) s += x8[e] * bf2f(w[e]);
        s = wave_reduce(s);
        if (lane == 0) {
            s += bias[row];
            out[row] = relu ? (s > 0.0f ? s : 0.0f) : s;
        }
    }
}

// out[row] = dot(W[row,0:1024], xin) + bias[row], 32 rows per wave
__device__ __forceinline__ void wave_gemv_k1024(const bf16_t* __restrict__ W,
                                                const float* __restrict__ bias,
                                                const float* __restrict__ xin,
                                                float* __restrict__ out,
                                                int wave, int lane) {
    float x32[32];
#pragma unroll
    for (int e = 0; e < 32; ++e) x32[e] = xin[lane * 32 + e];
    for (int r = 0; r < 32; ++r) {
        int row = wave * 32 + r;
        const v8bf* wp = (const v8bf*)(W + (size_t)row * 1024) + lane * 4;
        float s = 0.0f;
#pragma unroll
        for (int b4 = 0; b4 < 4; ++b4) {
            v8bf w = wp[b4];
#pragma unroll
            for (int e = 0; e < 8; ++e) s += x32[b4 * 8 + e] * bf2f(w[e]);
        }
        s = wave_reduce(s);
        if (lane == 0) out[row] = s + bias[row];
    }
}

__device__ __forceinline__ float block_sum(float v, float* red) {
    red[threadIdx.x] = v;
    __syncthreads();
    for (int s = 128; s > 0; s >>= 1) {
        if ((int)threadIdx.x < s) red[threadIdx.x] += red[threadIdx.x + s];
        __syncthreads();
    }
    float r = red[0];
    __syncthreads();
    return r;
}

__global__ void decode_kernel(DecArgs A) {
    __shared__ float q[256], tmp[256], ov[256], red[256];
    __shared__ float sc[8 * 200];
    __shared__ float ffn[1024];
    __shared__ float distA[64], distM[256], distP[64];
    __shared__ float hmax[8], hsum[8];
    __shared__ int   sel[200];

    const int b = blockIdx.x, tid = threadIdx.x;
    const int lane = tid & 31, wave = tid >> 5;
    q[tid] = A.q0[b * DMODEL + tid];
    __syncthreads();

    for (int t = 0; t < TSTEPS; ++t) {
        // ---- squared distances ----
        const float* ap = A.agent_poses + ((size_t)b * TSTEPS + t) * 64 * 2;
        const float* mp = A.map_poses   + ((size_t)b * TSTEPS + t) * 256 * 2;
        const float* pp = A.poly_poses  + ((size_t)b * TSTEPS + t) * 64 * 2;
        if (tid < 64) { float x = ap[tid * 2], y = ap[tid * 2 + 1]; distA[tid] = x * x + y * y; }
        { float x = mp[tid * 2], y = mp[tid * 2 + 1]; distM[tid] = x * x + y * y; }
        if (tid < 64) { float x = pp[tid * 2], y = pp[tid * 2 + 1]; distP[tid] = x * x + y * y; }
        __syncthreads();

        // ---- stable-rank top-k selection ----
        if (tid < 64) {
            float di = distA[tid]; int rk = 0;
            for (int j = 0; j < 64; ++j) { float dj = distA[j]; rk += (dj < di) || (dj == di && j < tid); }
            if (rk < 32) sel[rk] = SEG_AG + tid;
        }
        {
            float di = distM[tid]; int rk = 0;
            for (int j = 0; j < 256; ++j) { float dj = distM[j]; rk += (dj < di) || (dj == di && j < tid); }
            if (rk < 128) sel[32 + rk] = SEG_LN + tid;
        }
        if (tid < 8) sel[160 + tid] = SEG_RT + tid;
        if (tid < 64) {
            float di = distP[tid]; int rk = 0;
            for (int j = 0; j < 64; ++j) { float dj = distP[j]; rk += (dj < di) || (dj == di && j < tid); }
            if (rk < 32) sel[168 + rk] = SEG_PL + tid;
        }
        __syncthreads();

        // ---- 3 decoder layers ----
        for (int l = 0; l < 3; ++l) {
            // Q projection (wave-GEMV, bf16 weights)
            wave_gemv256(A.wq[l], A.in_b[l], q, tmp, 32, wave, lane, false);
            __syncthreads();

            // scores: 8 heads x 200 keys, dh=32, K rows via v8bf loads
            const bf16_t* Kb = A.Kl[l] + (size_t)b * NTOK * DMODEL;
            for (int idx = tid; idx < 1600; idx += 256) {
                int h = idx / 200, j = idx - h * 200;
                const v8bf* kr = (const v8bf*)(Kb + (size_t)sel[j] * DMODEL + h * 32);
                float s = 0.0f;
#pragma unroll
                for (int b4 = 0; b4 < 4; ++b4) {
                    v8bf kv = kr[b4];
#pragma unroll
                    for (int e = 0; e < 8; ++e) s += tmp[h * 32 + b4 * 8 + e] * bf2f(kv[e]);
                }
                sc[h * 200 + j] = s * 0.17677669529663687f;   // 1/sqrt(32)
            }
            __syncthreads();
            if (tid < 8) {
                float mx = -1e30f;
                for (int j = 0; j < 200; ++j) mx = fmaxf(mx, sc[tid * 200 + j]);
                hmax[tid] = mx;
            }
            __syncthreads();
            for (int idx = tid; idx < 1600; idx += 256)
                sc[idx] = __expf(sc[idx] - hmax[idx / 200]);
            __syncthreads();
            if (tid < 8) {
                float sm = 0.0f;
                for (int j = 0; j < 200; ++j) sm += sc[tid * 200 + j];
                hsum[tid] = sm;
            }
            __syncthreads();

            // attn-weighted V (coalesced across threads)
            {
                const bf16_t* Vb = A.Vl[l] + (size_t)b * NTOK * DMODEL;
                int h = tid >> 5;
                float s = 0.0f;
                for (int j = 0; j < 200; ++j)
                    s += sc[h * 200 + j] * bf2f(Vb[(size_t)sel[j] * DMODEL + tid]);
                ov[tid] = s / hsum[h];
            }
            __syncthreads();

            // out proj (wave-GEMV) + residual + LN1
            wave_gemv256(A.wo[l], A.out_b[l], ov, tmp, 32, wave, lane, false);
            __syncthreads();
            {
                float x = q[tid] + tmp[tid];
                float mean = block_sum(x, red) * (1.0f / 256.0f);
                float dv = x - mean;
                float var = block_sum(dv * dv, red) * (1.0f / 256.0f);
                q[tid] = dv * rsqrtf(var + 1e-5f) * A.g1[l][tid] + A.bb1[l][tid];
            }
            __syncthreads();

            // FFN: 256 -> 1024 (relu) -> 256, residual + LN2
            wave_gemv256(A.f1w[l], A.fb1[l], q, ffn, 128, wave, lane, true);
            __syncthreads();
            wave_gemv_k1024(A.f2w[l], A.fb2[l], ffn, tmp, wave, lane);
            __syncthreads();
            {
                float x = q[tid] + tmp[tid];
                float mean = block_sum(x, red) * (1.0f / 256.0f);
                float dv = x - mean;
                float var = block_sum(dv * dv, red) * (1.0f / 256.0f);
                q[tid] = dv * rsqrtf(var + 1e-5f) * A.g2[l][tid] + A.bb2[l][tid];
            }
            __syncthreads();
        }

        // ---- action head (wave 0, 2 rows, fp32 weights) ----
        if (wave == 0) {
            float x8[8];
#pragma unroll
            for (int e = 0; e < 8; ++e) x8[e] = q[lane * 8 + e];
#pragma unroll
            for (int row = 0; row < 2; ++row) {
                const float4* wr = (const float4*)(A.act_w + row * 256) + lane * 2;
                float4 w0 = wr[0], w1 = wr[1];
                float s = x8[0] * w0.x + x8[1] * w0.y + x8[2] * w0.z + x8[3] * w0.w
                        + x8[4] * w1.x + x8[5] * w1.y + x8[6] * w1.z + x8[7] * w1.w;
                s = wave_reduce(s);
                if (lane == 0)
                    A.out[((size_t)b * TSTEPS + t) * 2 + row] = s + A.act_b[row];
            }
        }
        __syncthreads();
    }
}

// ---------------------------------------------------------------------------
// Host launch
// ---------------------------------------------------------------------------
extern "C" void kernel_launch(void* const* d_in, const int* in_sizes, int n_in,
                              void* d_out, int out_size, void* d_ws, size_t ws_size,
                              hipStream_t stream) {
    (void)in_sizes; (void)n_in; (void)out_size; (void)ws_size;

    const float* agents      = (const float*)d_in[0];
    const float* agent_mask  = (const float*)d_in[1];
    const float* lanes       = (const float*)d_in[2];
    const float* lane_mask   = (const float*)d_in[3];
    const float* route       = (const float*)d_in[4];
    const float* route_mask  = (const float*)d_in[5];
    const float* polys       = (const float*)d_in[6];
    const float* poly_mask   = (const float*)d_in[7];
    const float* agent_poses = (const float*)d_in[8];
    const float* map_poses   = (const float*)d_in[9];
    const float* poly_poses  = (const float*)d_in[10];
    const int*   mode_c      = (const int*)d_in[11];
    const float* ea[6]; const float* el[6]; const float* er[6]; const float* ep[6];
    for (int i = 0; i < 6; ++i) {
        ea[i] = (const float*)d_in[12 + i];
        el[i] = (const float*)d_in[18 + i];
        er[i] = (const float*)d_in[24 + i];
        ep[i] = (const float*)d_in[30 + i];
    }
    const float* combine_w = (const float*)d_in[36];
    const float* combine_b = (const float*)d_in[37];
    const float* act_w = (const float*)d_in[74];
    const float* act_b = (const float*)d_in[75];

    // workspace layout
    const size_t FEATS = (size_t)BATCH * NTOK * DMODEL;   // fp32 elements
    float*  feats = (float*)d_ws;
    float*  q0    = feats + FEATS;
    bf16_t* kvb   = (bf16_t*)(q0 + (size_t)BATCH * DMODEL);
    bf16_t* decw  = kvb + 6 * FEATS;                      // bf16 decoder weights

    DecArgs A;
    A.agent_poses = agent_poses; A.map_poses = map_poses; A.poly_poses = poly_poses;
    A.q0 = q0; A.act_w = act_w; A.act_b = act_b; A.out = (float*)d_out;

    const float* in_w_f32[3]; const float* out_w_f32[3];
    const float* f1_f32[3];   const float* f2_f32[3];
    bf16_t* wcur = decw;
    for (int l = 0; l < 3; ++l) {
        int base = 38 + 12 * l;
        in_w_f32[l]  = (const float*)d_in[base + 0];
        A.in_b[l]    = (const float*)d_in[base + 1];
        out_w_f32[l] = (const float*)d_in[base + 2];
        A.out_b[l]   = (const float*)d_in[base + 3];
        A.g1[l]      = (const float*)d_in[base + 4];
        A.bb1[l]     = (const float*)d_in[base + 5];
        f1_f32[l]    = (const float*)d_in[base + 6];
        A.fb1[l]     = (const float*)d_in[base + 7];
        f2_f32[l]    = (const float*)d_in[base + 8];
        A.fb2[l]     = (const float*)d_in[base + 9];
        A.g2[l]      = (const float*)d_in[base + 10];
        A.bb2[l]     = (const float*)d_in[base + 11];
        A.Kl[l] = kvb + (size_t)(2 * l)     * FEATS;
        A.Vl[l] = kvb + (size_t)(2 * l + 1) * FEATS;
        A.wq[l]  = wcur; wcur += 256 * 256;
        A.wo[l]  = wcur; wcur += 256 * 256;
        A.f1w[l] = wcur; wcur += 1024 * 256;
        A.f2w[l] = wcur; wcur += 256 * 1024;
    }

    // --- encoders -> feats_all (B, 392, 256) ---
    enc_kernel<<<BATCH * 64 / 32, 256, 0, stream>>>(agents, agent_mask,
        ea[0], ea[1], ea[2], ea[3], ea[4], ea[5], feats, 4, 1, 64, SEG_AG);
    enc_kernel<<<BATCH * 256, 256, 0, stream>>>(lanes, lane_mask,
        el[0], el[1], el[2], el[3], el[4], el[5], feats, 3, 20, 256, SEG_LN);
    enc_kernel<<<BATCH * 8, 256, 0, stream>>>(route, route_mask,
        er[0], er[1], er[2], er[3], er[4], er[5], feats, 9, 20, 8, SEG_RT);
    enc_kernel<<<BATCH * 64, 256, 0, stream>>>(polys, poly_mask,
        ep[0], ep[1], ep[2], ep[3], ep[4], ep[5], feats, 3, 16, 64, SEG_PL);

    // --- query0 ---
    query0_kernel<<<BATCH, 256, 0, stream>>>(feats, combine_w, combine_b, mode_c, q0);

    // --- per-layer K/V precompute (WMMA) + decoder weight bf16 conversion ---
    const int M = BATCH * NTOK;                      // 25088
    const int gblocks = (M / 16) * (DMODEL / 16) / 8;
    for (int l = 0; l < 3; ++l) {
        gemm_xwt_kernel<<<gblocks, 256, 0, stream>>>(feats, in_w_f32[l] + 256 * 256,
                                                     A.in_b[l] + 256, (bf16_t*)A.Kl[l], M);
        gemm_xwt_kernel<<<gblocks, 256, 0, stream>>>(feats, in_w_f32[l] + 512 * 256,
                                                     A.in_b[l] + 512, (bf16_t*)A.Vl[l], M);
        cvt_bf16_kernel<<<(256 * 256 / 4 + 255) / 256, 256, 0, stream>>>(
            in_w_f32[l], (bf16_t*)A.wq[l], 256 * 256);
        cvt_bf16_kernel<<<(256 * 256 / 4 + 255) / 256, 256, 0, stream>>>(
            out_w_f32[l], (bf16_t*)A.wo[l], 256 * 256);
        cvt_bf16_kernel<<<(1024 * 256 / 4 + 255) / 256, 256, 0, stream>>>(
            f1_f32[l], (bf16_t*)A.f1w[l], 1024 * 256);
        cvt_bf16_kernel<<<(256 * 1024 / 4 + 255) / 256, 256, 0, stream>>>(
            f2_f32[l], (bf16_t*)A.f2w[l], 256 * 1024);
    }

    // --- autoregressive decode ---
    decode_kernel<<<BATCH, 256, 0, stream>>>(A);
}
